// DSGIM_26671746908788
// MI455X (gfx1250) — matrix operand order repaired and checked by reading, not compiled, not run on hardware
//
#include <hip/hip_runtime.h>

// CDNA5 / gfx1250 implementation. All GEMM-shaped work (q/k/v 1x1 convs, attn,
// w@V, fc, both 3x3 convs via implicit GEMM) runs through
// v_wmma_f32_16x16x32_f16 (f16 operands, f32 accumulate). GEMM staging uses
// gfx1250 async global->LDS copies (ASYNCcnt) with a double-buffered single
// barrier per K-step. Requires ws_size >= ~248 MB.

typedef __attribute__((ext_vector_type(16))) _Float16 v16h;
typedef __attribute__((ext_vector_type(8)))  float    v8f;

#define BM 128
#define BN 128
#define BK 32

static __device__ __forceinline__ v8f v8f_zero() {
  v8f z;
#pragma unroll
  for (int i = 0; i < 8; ++i) z[i] = 0.0f;
  return z;
}

// Async global->LDS 16-byte copy (CDNA5, tracked by ASYNCcnt).
static __device__ __forceinline__ void async_b128(_Float16* lds, const _Float16* g) {
  const unsigned l = (unsigned)(size_t)lds;               // low 32 bits = LDS offset
  const unsigned long long a = (unsigned long long)(size_t)g;
  asm volatile("global_load_async_to_lds_b128 %0, %1, off" :: "v"(l), "v"(a) : "memory");
}
static __device__ __forceinline__ void wait_async0() {
  asm volatile("s_wait_asynccnt 0x0" ::: "memory");
}

// A fragment: 16x32 f16 tile, row-major in LDS (32 halfs/row).
// Lane l (l<16) holds row M=l, K {0..7,16..23}; lane l+16 holds K {8..15,24..31}.
static __device__ __forceinline__ v16h frag_a(const _Float16* tile, int lane) {
  const int r = lane & 15, kh = lane >> 4;
  const unsigned int* row = (const unsigned int*)(tile + r * BK);
  union { unsigned int u[8]; v16h v; } f;
#pragma unroll
  for (int j = 0; j < 4; ++j) {
    f.u[j]     = row[kh * 4 + j];
    f.u[4 + j] = row[kh * 4 + 8 + j];
  }
  return f.v;
}

// B fragment from B^T tile (16 N-rows x 32 K-halfs row-major in LDS).
// Lane n (n<16) holds column N=n, K {0..7,16..23}; lane n+16 holds K {8..15,24..31}.
static __device__ __forceinline__ v16h frag_b(const _Float16* tile, int lane) {
  const int n = lane & 15, kh = lane >> 4;
  const unsigned int* row = (const unsigned int*)(tile + n * BK);
  union { unsigned int u[8]; v16h v; } f;
#pragma unroll
  for (int j = 0; j < 8; ++j) f.u[j] = row[kh * 4 + (j >> 2) * 8 + (j & 3)];
  return f.v;
}

// ---------------------------------------------------------------- utilities

__global__ void k_cvt(const float* s, _Float16* d, int n) {
  int i = blockIdx.x * blockDim.x + threadIdx.x;
  if (i < n) d[i] = (_Float16)s[i];
}

// ------------------------------------------------------- window select/mask
// ranking-equivalent: sum(triu(cos)) = 0.5*(||sum_i a_i/|a_i|||^2 + 49)
__global__ void __launch_bounds__(256) k_win_score(const float* x, float* ss) {
  __shared__ float pn[49][4];
  __shared__ float rn[49];
  __shared__ float red[256];
  const int b = blockIdx.x >> 2, wd = blockIdx.x & 3;
  const int h0 = (wd >> 1) * 7, w0 = (wd & 1) * 7;
  const int tid = threadIdx.x;
  if (tid < 196) {
    const int i = tid >> 2, qc = tid & 3;
    const int p = (h0 + i / 7) * 14 + (w0 + i % 7);
    float s = 0.0f;
    for (int c = qc * 96; c < qc * 96 + 96; ++c) {
      float v = x[((long long)b * 384 + c) * 196 + p];
      s += v * v;
    }
    pn[i][qc] = s;
  }
  __syncthreads();
  if (tid < 49)
    rn[tid] = rsqrtf(fmaxf(pn[tid][0] + pn[tid][1] + pn[tid][2] + pn[tid][3], 1e-16f));
  __syncthreads();
  float Sp = 0.0f;
  for (int c = tid; c < 384; c += 256) {
    float sc = 0.0f;
    for (int i = 0; i < 49; ++i) {
      const int p = (h0 + i / 7) * 14 + (w0 + i % 7);
      sc += x[((long long)b * 384 + c) * 196 + p] * rn[i];
    }
    Sp += sc * sc;
  }
  red[tid] = Sp;
  __syncthreads();
  for (int off = 128; off; off >>= 1) {
    if (tid < off) red[tid] += red[tid + off];
    __syncthreads();
  }
  if (tid == 0) ss[blockIdx.x] = red[0];
}

__global__ void k_mask(const float* ss, const int* gamm, float* mq, float* mk) {
  int gid = blockIdx.x * blockDim.x + threadIdx.x;
  if (gid >= 256 * 196) return;
  const int b = gid / 196, p = gid - b * 196;
  float sv[4];
  for (int t = 0; t < 4; ++t) sv[t] = ss[b * 4 + t];
  int i1 = 0;
  for (int t = 1; t < 4; ++t) if (sv[t] > sv[i1]) i1 = t;       // top-k tie: lowest idx
  int i2 = -1;
  for (int t = 0; t < 4; ++t) {
    if (t == i1) continue;
    if (i2 < 0 || sv[t] > sv[i2]) i2 = t;
  }
  const int h = p / 14, wcol = p - h * 14;
  const int wid = (h / 7) * 2 + (wcol / 7);
  const int member = (wid == i1 || wid == i2) ? 1 : 0;
  float a = member ? 0.0f : 1.0f, bq = 1.0f - a;
  if (b == 0) { float g = (float)(*gamm); a = g; bq = g; }
  mq[gid] = a;
  mk[gid] = bq;
}

// ------------------------------------------- x (B,C,196) -> masked f16 rows
__global__ void __launch_bounds__(256) k_pack_x(const float* x, const float* mq, const float* mk,
                                                _Float16* xq, _Float16* xk, _Float16* xv) {
  __shared__ float tile[32][64 + 1];
  const int b = blockIdx.x, c0 = blockIdx.y * 32, p0 = blockIdx.z * 64;
  const int tid = threadIdx.x;
#pragma unroll
  for (int it = 0; it < 8; ++it) {
    const int idx = tid + it * 256;
    const int cl = idx >> 6, pl = idx & 63;
    const int p = p0 + pl;
    float v = 0.0f;
    if (p < 196) v = x[((long long)b * 384 + c0 + cl) * 196 + p];
    tile[cl][pl] = v;
  }
  __syncthreads();
#pragma unroll
  for (int it = 0; it < 8; ++it) {
    const int idx = tid + it * 256;
    const int cl = idx & 31, pl = idx >> 5;
    const int p = p0 + pl;
    if (p < 196) {
      const float v = tile[cl][pl];
      const float a = mq[b * 196 + p], bq = mk[b * 196 + p];
      const long long base = ((long long)b * 196 + p) * 384 + c0 + cl;
      xq[base] = (_Float16)(v * a);
      xk[base] = (_Float16)(v * bq);
      xv[base] = (_Float16)v;
    }
  }
}

// ---------------------------------------------------------- generic WMMA GEMM
// 128x128 block tile, 8 waves each owning a 16x128 strip (8 f32 accumulators).
// Double-buffered LDS, async global->LDS staging, one barrier per K-step.
struct GemmParams {
  const _Float16* A; long long sAb; int lda; int Mvalid; int Kvalid; int Ksteps;
  const _Float16* B; long long sBb; int ldb; int Nvalid;   // B^T layout (N rows, K cols)
  void* O; long long sOb; int ldo;
  const float* bias;
  int mode;   // 0 = f16 row-major, 1 = f32 row-major, 2 = f32 scatter (b,n,p) NCHW
  int rpi;    // rows per image (scatter)
};

__global__ void __launch_bounds__(256) k_gemm(GemmParams p) {
  __shared__ __align__(16) _Float16 sA[2][BM * BK];
  __shared__ __align__(16) _Float16 sB[2][BN * BK];
  const int tid = threadIdx.x, wave = tid >> 5, lane = tid & 31;
  const int bz = blockIdx.z;
  const int m0 = blockIdx.x * BM, n0 = blockIdx.y * BN;
  const _Float16* Ab = p.A + (long long)bz * p.sAb;
  const _Float16* Bb = p.B + (long long)bz * p.sBb;

  v8f acc[8];
#pragma unroll
  for (int t = 0; t < 8; ++t) acc[t] = v8f_zero();

  const int ra = tid >> 1, ha = (tid & 1) * 16;   // 128 rows x 32 halfs, 16 halfs/thread
  const int gma = m0 + ra, gnb = n0 + ra;

  auto stage = [&](int ks, int buf) {
    const int k0 = ks * BK;
    const bool fullk = (k0 + BK) <= p.Kvalid;
    _Float16* da = &sA[buf][ra * BK + ha];
    if (gma < p.Mvalid && fullk) {
      const _Float16* src = Ab + (long long)gma * p.lda + k0 + ha;
      async_b128(da, src);
      async_b128(da + 8, src + 8);
    } else {
#pragma unroll
      for (int e = 0; e < 16; ++e) {
        const int k = k0 + ha + e;
        da[e] = (gma < p.Mvalid && k < p.Kvalid) ? Ab[(long long)gma * p.lda + k]
                                                 : (_Float16)0.0f;
      }
    }
    _Float16* db = &sB[buf][ra * BK + ha];
    if (gnb < p.Nvalid && fullk) {
      const _Float16* src = Bb + (long long)gnb * p.ldb + k0 + ha;
      async_b128(db, src);
      async_b128(db + 8, src + 8);
    } else {
#pragma unroll
      for (int e = 0; e < 16; ++e) {
        const int k = k0 + ha + e;
        db[e] = (gnb < p.Nvalid && k < p.Kvalid) ? Bb[(long long)gnb * p.ldb + k]
                                                 : (_Float16)0.0f;
      }
    }
  };

  stage(0, 0);
  int cur = 0;
  for (int ks = 0; ks < p.Ksteps; ++ks) {
    wait_async0();        // own async copies into sX[cur] complete
    __syncthreads();      // all waves' copies visible; prev reads of sX[cur^1] done
    if (ks + 1 < p.Ksteps) stage(ks + 1, cur ^ 1);
    const v16h af = frag_a(&sA[cur][(wave * 16) * BK], lane);
#pragma unroll
    for (int t = 0; t < 8; ++t) {
      const v16h bf = frag_b(&sB[cur][(t * 16) * BK], lane);
      acc[t] = __builtin_amdgcn_wmma_f32_16x16x32_f16(false, af, false, bf,
                                                      (short)0, acc[t], false, false);
    }
    cur ^= 1;
  }

  const int mm = m0 + wave * 16 + (lane >> 4) * 8;
  const int nb = n0 + (lane & 15);
#pragma unroll
  for (int t = 0; t < 8; ++t) {
    const int n = nb + t * 16;
    if (n >= p.Nvalid) continue;
    const float bv = p.bias ? p.bias[n] : 0.0f;
#pragma unroll
    for (int j = 0; j < 8; ++j) {
      const int m = mm + j;
      if (m >= p.Mvalid) continue;
      const float v = acc[t][j] + bv;
      if (p.mode == 0)
        ((_Float16*)p.O)[(long long)bz * p.sOb + (long long)m * p.ldo + n] = (_Float16)v;
      else if (p.mode == 1)
        ((float*)p.O)[(long long)bz * p.sOb + (long long)m * p.ldo + n] = v;
      else {
        const int b2 = m / p.rpi, pp = m - b2 * p.rpi;
        ((float*)p.O)[((long long)b2 * p.Nvalid + n) * p.rpi + pp] = v;
      }
    }
  }
}

// ------------------------------------------------- spa + 98th-largest select
__global__ void __launch_bounds__(256) k_spa_last(const float* attn, float* spaG, float* lastG) {
  __shared__ float sspa[196];
  const int b = blockIdx.x, i = threadIdx.x;
  if (i < 196) {
    const float* row = attn + ((long long)b * 196 + i) * 196;
    float s = 0.0f;
    for (int j = 0; j < 196; ++j) s += row[j];
    sspa[i] = s * (1.0f / 196.0f);
  }
  __syncthreads();
  if (i < 196) {
    const float si = sspa[i];
    float last = 3.4e38f;
    for (int j = 0; j < 196; ++j) {
      const float d = fabsf(si - sspa[j]);
      int cnt = 0;
      for (int j2 = 0; j2 < 196; ++j2) cnt += (fabsf(si - sspa[j2]) > d) ? 1 : 0;
      if (cnt <= 97 && d < last) last = d;   // 98th largest = min value with rank<=98
    }
    spaG[b * 196 + i] = si;
    lastG[b * 196 + i] = last;
  }
}

// ---------------------------------------------- masked double softmax -> f16
__global__ void __launch_bounds__(256) k_weight(const float* attn, const float* spaG,
                                                const float* lastG, _Float16* wmat) {
  __shared__ float sspa[196];
  const int b = blockIdx.y, tid = threadIdx.x;
  if (tid < 196) sspa[tid] = spaG[b * 196 + tid];
  __syncthreads();
  const int wave = tid >> 5, lane = tid & 31;
  const int i = blockIdx.x * 8 + wave;
  if (i >= 196) return;
  const float si = sspa[i];
  const float lasti = lastG[b * 196 + i];
  const float* arow = attn + ((long long)b * 196 + i) * 196;
  float a1v[7], a2v[7];
  float m1 = -1e30f, m2 = -1e30f;
#pragma unroll
  for (int cc = 0; cc < 7; ++cc) {
    const int j = cc * 32 + lane;
    float a1 = -1e30f, a2 = -1e30f;
    if (j < 196) {
      const float att = arow[j];
      const float d = fabsf(si - sspa[j]) + ((i == j) ? 10000.0f : 0.0f);
      const bool os1 = (d >= lasti);
      const bool os2 = ((os1 ? 1.0f : 0.0f) <= lasti);
      a1 = (os1 && att != 0.0f) ? att : -100.0f;
      a2 = (os2 && att != 0.0f) ? att : -100.0f;
    }
    a1v[cc] = a1; a2v[cc] = a2;
    m1 = fmaxf(m1, a1); m2 = fmaxf(m2, a2);
  }
  for (int off = 16; off; off >>= 1) {
    m1 = fmaxf(m1, __shfl_xor(m1, off));
    m2 = fmaxf(m2, __shfl_xor(m2, off));
  }
  float s1 = 0.0f, s2 = 0.0f;
#pragma unroll
  for (int cc = 0; cc < 7; ++cc) {
    const int j = cc * 32 + lane;
    if (j < 196) { s1 += __expf(a1v[cc] - m1); s2 += __expf(a2v[cc] - m2); }
  }
  for (int off = 16; off; off >>= 1) {
    s1 += __shfl_xor(s1, off);
    s2 += __shfl_xor(s2, off);
  }
  const float r1 = 1.0f / s1, r2 = 1.0f / s2;
#pragma unroll
  for (int cc = 0; cc < 7; ++cc) {
    const int j = cc * 32 + lane;
    if (j < 196) {
      const float wv = __expf(a1v[cc] - m1) * r1 + __expf(a2v[cc] - m2) * r2;
      wmat[((long long)b * 196 + i) * 224 + j] = (_Float16)wv;
    }
  }
}

// ---------------------------------------------- V (b,196,384) -> V^T (b,384,224)
__global__ void __launch_bounds__(256) k_transpose_v(const _Float16* V, _Float16* VT) {
  __shared__ _Float16 tile[32][64 + 2];
  const int b = blockIdx.x, c0 = blockIdx.y * 32, j0 = blockIdx.z * 64;
  const int tid = threadIdx.x;
#pragma unroll
  for (int it = 0; it < 8; ++it) {
    const int idx = tid + it * 256;
    const int jl = idx >> 5, cl = idx & 31;
    const int j = j0 + jl;
    _Float16 v = (_Float16)0.0f;
    if (j < 196) v = V[((long long)b * 196 + j) * 384 + c0 + cl];
    tile[cl][jl] = v;
  }
  __syncthreads();
#pragma unroll
  for (int it = 0; it < 8; ++it) {
    const int idx = tid + it * 256;
    const int cl = idx >> 6, jl = idx & 63;
    const int j = j0 + jl;
    if (j < 196) VT[((long long)b * 384 + c0 + cl) * 224 + j] = tile[cl][jl];
  }
}

// --------------------------------------------- 3x3 conv as implicit WMMA GEMM
struct ConvParams {
  const float* In; int Cin;                 // (B, Cin, 196) f32
  const float* scale; const float* shift;   // fused input BN+ReLU (null = none)
  const _Float16* W; int N; int ldw; int Ksteps;  // weights (N, Cin*9) f16
  float* O; const float* bias;              // (B, N, 196) f32
};

__global__ void __launch_bounds__(256) k_conv(ConvParams p) {
  __shared__ __align__(16) _Float16 sA[BM * BK];
  __shared__ __align__(16) _Float16 sB[BN * BK];
  const int tid = threadIdx.x, wave = tid >> 5, lane = tid & 31;
  const int m0 = blockIdx.x * BM, n0 = blockIdx.y * BN;

  const int ra = tid >> 1, ha = (tid & 1) * 16;
  const int gm = m0 + ra;
  const int bimg = gm / 196, pp = gm - bimg * 196;
  const int h2 = pp / 14, w2 = pp - h2 * 14;
  const float* inb = p.In + (long long)bimg * p.Cin * 196;
  const int gn = n0 + ra;

  v8f acc[8];
#pragma unroll
  for (int t = 0; t < 8; ++t) acc[t] = v8f_zero();

  for (int ks = 0; ks < p.Ksteps; ++ks) {
    const int k0 = ks * BK;
    // weight tile: async global->LDS (contiguous rows of the f16 weight matrix)
    _Float16* db = sB + ra * BK + ha;
    if (gn < p.N) {
      const _Float16* src = p.W + (long long)gn * p.ldw + k0 + ha;
      async_b128(db, src);
      async_b128(db + 8, src + 8);
    } else {
#pragma unroll
      for (int e = 0; e < 16; ++e) db[e] = (_Float16)0.0f;
    }
    // im2col gather (f32 -> f16, optional fused BN+ReLU) via VGPRs
    _Float16* da = sA + ra * BK + ha;
#pragma unroll
    for (int e = 0; e < 16; ++e) {
      const int k = k0 + ha + e;
      const int c = k / 9, t9 = k - c * 9;
      const int dy = t9 / 3 - 1, dx = (t9 % 3) - 1;
      const int hy = h2 + dy, wx = w2 + dx;
      float v = 0.0f;
      if (hy >= 0 && hy < 14 && wx >= 0 && wx < 14) {
        v = inb[c * 196 + hy * 14 + wx];
        if (p.scale) v = fmaxf(v * p.scale[c] + p.shift[c], 0.0f);
      }
      da[e] = (_Float16)v;
    }
    wait_async0();
    __syncthreads();
    const v16h af = frag_a(sA + (wave * 16) * BK, lane);
#pragma unroll
    for (int t = 0; t < 8; ++t) {
      const v16h bf = frag_b(sB + (t * 16) * BK, lane);
      acc[t] = __builtin_amdgcn_wmma_f32_16x16x32_f16(false, af, false, bf,
                                                      (short)0, acc[t], false, false);
    }
    __syncthreads();
  }

  const int mm = m0 + wave * 16 + (lane >> 4) * 8;
  const int nb = n0 + (lane & 15);
#pragma unroll
  for (int t = 0; t < 8; ++t) {
    const int n = nb + t * 16;
    if (n >= p.N) continue;
    const float bv = p.bias ? p.bias[n] : 0.0f;
#pragma unroll
    for (int j = 0; j < 8; ++j) {
      const int m = mm + j;
      const int b2 = m / 196, pq = m - b2 * 196;
      p.O[((long long)b2 * p.N + n) * 196 + pq] = acc[t][j] + bv;
    }
  }
}

// ------------------------------------------------------------- BN (train)
__global__ void __launch_bounds__(256) k_bn_stats(const float* X, int C, const float* g,
                                                  const float* beta, float* scale, float* shift) {
  __shared__ float rs[256], rq[256];
  const int c = blockIdx.x, tid = threadIdx.x;
  float s = 0.0f, q = 0.0f;
  const int n = 256 * 196;
  for (int idx = tid; idx < n; idx += 256) {
    const int b = idx / 196, pp = idx - b * 196;
    const float v = X[((long long)b * C + c) * 196 + pp];
    s += v; q += v * v;
  }
  rs[tid] = s; rq[tid] = q;
  __syncthreads();
  for (int off = 128; off; off >>= 1) {
    if (tid < off) { rs[tid] += rs[tid + off]; rq[tid] += rq[tid + off]; }
    __syncthreads();
  }
  if (tid == 0) {
    const float mean = rs[0] / n;
    const float var = rq[0] / n - mean * mean;
    const float sc = g[c] * rsqrtf(var + 1e-5f);
    scale[c] = sc;
    shift[c] = beta[c] - mean * sc;
  }
}

__global__ void k_bn_relu_out(const float* y, const float* scale, const float* shift,
                              float* out, int total) {
  const int idx = blockIdx.x * blockDim.x + threadIdx.x;
  if (idx >= total) return;
  const int c = (idx / 196) % 384;
  out[idx] = fmaxf(y[idx] * scale[c] + shift[c], 0.0f);
}

// ---------------------------------------------------------------- launcher
extern "C" void kernel_launch(void* const* d_in, const int* in_sizes, int n_in,
                              void* d_out, int out_size, void* d_ws, size_t ws_size,
                              hipStream_t stream) {
  (void)in_sizes; (void)n_in; (void)out_size; (void)ws_size;
  const float* x    = (const float*)d_in[0];
  const float* qw   = (const float*)d_in[1];
  const float* qbv  = (const float*)d_in[2];
  const float* kw   = (const float*)d_in[3];
  const float* kbv  = (const float*)d_in[4];
  const float* vw   = (const float*)d_in[5];
  const float* vbv  = (const float*)d_in[6];
  const float* fcw  = (const float*)d_in[7];
  const float* fcb  = (const float*)d_in[8];
  const float* c1w  = (const float*)d_in[9];
  const float* c1b  = (const float*)d_in[10];
  const float* bn1g = (const float*)d_in[11];
  const float* bn1b = (const float*)d_in[12];
  const float* c2w  = (const float*)d_in[13];
  const float* c2b  = (const float*)d_in[14];
  const float* bn2g = (const float*)d_in[15];
  const float* bn2b = (const float*)d_in[16];
  const int*   gamm = (const int*)d_in[17];

  char* w = (char*)d_ws;
  // small region
  float* ss   = (float*)(w + 0);
  float* mq   = (float*)(w + 4096);
  float* mk   = (float*)(w + 204800);
  float* spa  = (float*)(w + 405504);
  float* last = (float*)(w + 606208);
  float* sc1  = (float*)(w + 806912);
  float* sh1  = (float*)(w + 807296);
  float* sc2  = (float*)(w + 807680);
  float* sh2  = (float*)(w + 809216);
  _Float16* Wq  = (_Float16*)(w + 810752);
  _Float16* Wk  = (_Float16*)(w + 1105664);
  _Float16* Wv  = (_Float16*)(w + 1400576);
  _Float16* Wfc = (_Float16*)(w + 1695488);
  _Float16* Wc1 = (_Float16*)(w + 1990400);
  _Float16* Wc2 = (_Float16*)(w + 2653952);

  const size_t AA  = (size_t)4 << 20;                 // arena A (116 MB, reused)
  const size_t SZX = 38535168ull;                     // 50176*384*2
  const size_t AB  = AA + ((size_t)116 << 20);        // arena B (116 MB, reused)
  _Float16* xq   = (_Float16*)(w + AA);
  _Float16* xk   = (_Float16*)(w + AA + SZX);
  _Float16* xv   = (_Float16*)(w + AA + 2 * SZX);
  float*    attn = (float*)(w + AA);                  // reuses xq..xv (dead)
  _Float16* wm   = (_Float16*)(w + AA + ((size_t)40 << 20));
  _Float16* vt   = (_Float16*)(w + AA + ((size_t)64 << 20));
  _Float16* feat = (_Float16*)(w + AA);               // reuses attn (dead)
  _Float16* Qm   = (_Float16*)(w + AB);
  _Float16* Km   = (_Float16*)(w + AB + SZX);
  _Float16* Vm   = (_Float16*)(w + AB + 2 * SZX);
  float*    y1   = (float*)(w + AB);                  // reuses Qm/Km (dead)
  float*    r1   = (float*)(w + AB + ((size_t)80 << 20));
  float*    y2   = (float*)(w + AB);                  // reuses y1 (dead)

  // 1) weights -> f16 (B^T = native (out, in) layout)
  k_cvt<<<(147456 + 255) / 256, 256, 0, stream>>>(qw, Wq, 147456);
  k_cvt<<<(147456 + 255) / 256, 256, 0, stream>>>(kw, Wk, 147456);
  k_cvt<<<(147456 + 255) / 256, 256, 0, stream>>>(vw, Wv, 147456);
  k_cvt<<<(147456 + 255) / 256, 256, 0, stream>>>(fcw, Wfc, 147456);
  k_cvt<<<(331776 + 255) / 256, 256, 0, stream>>>(c1w, Wc1, 331776);
  k_cvt<<<(331776 + 255) / 256, 256, 0, stream>>>(c2w, Wc2, 331776);

  // 2) window scores + per-pixel masks
  k_win_score<<<1024, 256, 0, stream>>>(x, ss);
  k_mask<<<(50176 + 255) / 256, 256, 0, stream>>>(ss, gamm, mq, mk);

  // 3) x -> masked f16 rows (b*196, 384)
  k_pack_x<<<dim3(256, 12, 4), 256, 0, stream>>>(x, mq, mk, xq, xk, xv);

  // 4) q/k/v GEMMs: M=50176, K=N=384
  GemmParams gp{};
  gp.sAb = 0; gp.lda = 384; gp.Mvalid = 50176; gp.Kvalid = 384; gp.Ksteps = 12;
  gp.sBb = 0; gp.ldb = 384; gp.Nvalid = 384;
  gp.sOb = 0; gp.ldo = 384; gp.mode = 0; gp.rpi = 196;
  const dim3 gBig(392, 3, 1);
  gp.A = xq; gp.B = Wq; gp.O = Qm; gp.bias = qbv; k_gemm<<<gBig, 256, 0, stream>>>(gp);
  gp.A = xk; gp.B = Wk; gp.O = Km; gp.bias = kbv; k_gemm<<<gBig, 256, 0, stream>>>(gp);
  gp.A = xv; gp.B = Wv; gp.O = Vm; gp.bias = vbv; k_gemm<<<gBig, 256, 0, stream>>>(gp);

  // 5) attn = Q @ K^T per batch (196x196, K=384)
  GemmParams ga{};
  ga.A = Qm; ga.sAb = 196 * 384; ga.lda = 384; ga.Mvalid = 196; ga.Kvalid = 384; ga.Ksteps = 12;
  ga.B = Km; ga.sBb = 196 * 384; ga.ldb = 384; ga.Nvalid = 196;
  ga.O = attn; ga.sOb = 196 * 196; ga.ldo = 196; ga.bias = nullptr; ga.mode = 1; ga.rpi = 196;
  k_gemm<<<dim3(2, 2, 256), 256, 0, stream>>>(ga);

  // 6) row means + 98th-largest threshold; 7) double softmax weights -> f16
  k_spa_last<<<256, 256, 0, stream>>>(attn, spa, last);
  k_weight<<<dim3(25, 256), 256, 0, stream>>>(attn, spa, last, wm);

  // 8) V^T for the w@V GEMM
  k_transpose_v<<<dim3(256, 12, 4), 256, 0, stream>>>(Vm, vt);

  // 9) feat = w @ V per batch (196x384, K=196 padded to 224)
  GemmParams gf{};
  gf.A = wm; gf.sAb = 196 * 224; gf.lda = 224; gf.Mvalid = 196; gf.Kvalid = 196; gf.Ksteps = 7;
  gf.B = vt; gf.sBb = 384 * 224; gf.ldb = 224; gf.Nvalid = 384;
  gf.O = feat; gf.sOb = 196 * 384; gf.ldo = 384; gf.bias = nullptr; gf.mode = 0; gf.rpi = 196;
  k_gemm<<<dim3(2, 3, 256), 256, 0, stream>>>(gf);

  // 10) fc -> y1 in NCHW (B,384,196) f32
  GemmParams gc{};
  gc.A = feat; gc.sAb = 0; gc.lda = 384; gc.Mvalid = 50176; gc.Kvalid = 384; gc.Ksteps = 12;
  gc.B = Wfc; gc.sBb = 0; gc.ldb = 384; gc.Nvalid = 384;
  gc.O = y1; gc.sOb = 0; gc.ldo = 0; gc.bias = fcb; gc.mode = 2; gc.rpi = 196;
  k_gemm<<<gBig, 256, 0, stream>>>(gc);

  // 11) conv1 (implicit GEMM, K=3456, N=96) -> r1 pre-BN
  ConvParams c1{};
  c1.In = y1; c1.Cin = 384; c1.scale = nullptr; c1.shift = nullptr;
  c1.W = Wc1; c1.N = 96; c1.ldw = 3456; c1.Ksteps = 108; c1.O = r1; c1.bias = c1b;
  k_conv<<<dim3(392, 1, 1), 256, 0, stream>>>(c1);

  // 12) BN1 stats; 13) conv2 with fused BN1+ReLU on input (K=864, N=384)
  k_bn_stats<<<96, 256, 0, stream>>>(r1, 96, bn1g, bn1b, sc1, sh1);
  ConvParams c2{};
  c2.In = r1; c2.Cin = 96; c2.scale = sc1; c2.shift = sh1;
  c2.W = Wc2; c2.N = 384; c2.ldw = 864; c2.Ksteps = 27; c2.O = y2; c2.bias = c2b;
  k_conv<<<dim3(392, 3, 1), 256, 0, stream>>>(c2);

  // 14) BN2 stats + final BN+ReLU into d_out
  k_bn_stats<<<384, 256, 0, stream>>>(y2, 384, bn2g, bn2b, sc2, sh2);
  k_bn_relu_out<<<(19267584 + 255) / 256, 256, 0, stream>>>(y2, sc2, sh2,
                                                            (float*)d_out, 19267584);
}
